// lstm_56401510531657
// MI455X (gfx1250) — matrix operand order repaired and checked by reading, not compiled, not run on hardware
//
#include <hip/hip_runtime.h>

typedef unsigned short u16;
typedef unsigned int   u32;
typedef __attribute__((ext_vector_type(4)))  int    v4i;
typedef __attribute__((ext_vector_type(16))) __bf16 v16bf;
typedef __attribute__((ext_vector_type(8)))  float  v8f;

struct alignas(16) U128 { u32 w[4]; };
union V16BF { U128 q[2]; v16bf v; };

constexpr int CB = 256;   // batch
constexpr int CT = 512;   // time
constexpr int CE = 256;   // embed
constexpr int CH = 256;   // hidden
constexpr int CV = 512;   // vocab
constexpr int CD = 768;   // E + 2H
constexpr int CW = 512;   // E + H (Wc K-dim)
constexpr int NWG = 32;   // persistent workgroups for the scan

__device__ __forceinline__ u16 f2bf(float f) {
  u32 u = __float_as_uint(f);
  u32 r = u + 0x7FFFu + ((u >> 16) & 1u);   // round-to-nearest-even
  return (u16)(r >> 16);
}
__device__ __forceinline__ float sigm(float x) {
  return 1.0f / (1.0f + __expf(-x));
}
__device__ __forceinline__ v8f wmma_bf16(v16bf a, v16bf b, v8f c) {
  return __builtin_amdgcn_wmma_f32_16x16x32_bf16(false, a, false, b, (short)0, c,
                                                 false, false);
}

// 16-byte global -> LDS stage; async-to-LDS path when the toolchain has it.
#if __has_builtin(__builtin_amdgcn_global_load_async_to_lds_b128)
#define HAVE_ASYNC_LDS 1
__device__ __forceinline__ void stage16(const u16* g, u16* l) {
  __builtin_amdgcn_global_load_async_to_lds_b128((v4i*)g, (v4i*)l, 0, 0);
}
__device__ __forceinline__ void stage_wait() {
#if __has_builtin(__builtin_amdgcn_s_wait_asynccnt)
  __builtin_amdgcn_s_wait_asynccnt(0);
#else
  asm volatile("s_wait_asynccnt 0" ::: "memory");
#endif
}
#else
#define HAVE_ASYNC_LDS 0
__device__ __forceinline__ void stage16(const u16* g, u16* l) {
  *(U128*)l = *(const U128*)g;
}
__device__ __forceinline__ void stage_wait() {}
#endif

// Device-wide barrier for the persistent scan kernel (monotonic counter).
__device__ __forceinline__ void grid_barrier(int* cnt, int target) {
  __threadfence();                    // release this WG's global stores
  __syncthreads();
  if (threadIdx.x == 0) {
    __hip_atomic_fetch_add(cnt, 1, __ATOMIC_RELEASE, __HIP_MEMORY_SCOPE_AGENT);
    while (__hip_atomic_load(cnt, __ATOMIC_ACQUIRE, __HIP_MEMORY_SCOPE_AGENT) < target)
      __builtin_amdgcn_s_sleep(2);
  }
  __syncthreads();
  __builtin_amdgcn_fence(__ATOMIC_ACQUIRE, "agent");
}

// ---------------------------------------------------------------------------
// Prep: transpose weights to [N][K] bf16, zero initial states + barrier.
// ---------------------------------------------------------------------------
__global__ void prep_weights(const float* __restrict__ Wf, const float* __restrict__ Wu,
                             const float* __restrict__ Wo, const float* __restrict__ Wc,
                             const float* __restrict__ Wy,
                             u16* __restrict__ WfT, u16* __restrict__ WuT,
                             u16* __restrict__ WoT, u16* __restrict__ WcT,
                             u16* __restrict__ WyT,
                             float* __restrict__ c32_0, u16* __restrict__ cbf_0,
                             u16* __restrict__ h0, int* __restrict__ bar) {
  int i = blockIdx.x * 256 + threadIdx.x;           // 0 .. 768*256-1
  if (i == 0) *bar = 0;
  if (i < CD * CH) {                                // Wf/Wu/Wo: [768][256] -> [256][768]
    int k = i / CH, n = i % CH;
    WfT[n * CD + k] = f2bf(Wf[i]);
    WuT[n * CD + k] = f2bf(Wu[i]);
    WoT[n * CD + k] = f2bf(Wo[i]);
  }
  if (i < CW * CH) {                                // Wc: [512][256] -> [256][512]
    int k = i / CH, n = i % CH;
    WcT[n * CW + k] = f2bf(Wc[i]);
  }
  if (i < CH * CV) {                                // Wy: [256][512] -> [512][256]
    int k = i / CV, n = i % CV;
    WyT[n * CH + k] = f2bf(Wy[i]);
  }
  if (i < CB * CH) {
    c32_0[i] = 0.0f;
    cbf_0[i] = 0;
    h0[i] = 0;
  }
}

// ---------------------------------------------------------------------------
// Embedding gather -> bf16 xe laid out [t][b][e] (time-major for the scan).
// ---------------------------------------------------------------------------
__global__ void embed_gather(const int* __restrict__ x, const float* __restrict__ embed,
                             u16* __restrict__ xe) {
  int bt = blockIdx.x;                // b*T + t
  int b = bt / CT, t = bt % CT;
  int e = threadIdx.x;                // 256 threads
  int tok = x[bt];
  xe[(size_t)t * CB * CE + (size_t)b * CE + e] = f2bf(embed[(size_t)tok * CE + e]);
}

// ---------------------------------------------------------------------------
// Persistent LSTM scan: one launch, 32 resident WGs, 512 steps inside.
// Each WG owns a 64-row x 32-col tile of the 256x256 (B x H) state and keeps
// its 32-column slab of Wf/Wu/Wo/Wc (176 KB bf16) in LDS for the whole scan.
// Waves: w&3 -> 16-row block, w>>2 -> 16-col block.
// ---------------------------------------------------------------------------
__global__ __launch_bounds__(256) void lstm_scan(
    float* __restrict__ c32a, float* __restrict__ c32b,
    u16* __restrict__ cbfa, u16* __restrict__ cbfb,
    const u16* __restrict__ h0, u16* __restrict__ hsbf,
    const u16* __restrict__ xe,
    const u16* __restrict__ WfT, const u16* __restrict__ WuT,
    const u16* __restrict__ WoT, const u16* __restrict__ WcT,
    const float* __restrict__ biasF, const float* __restrict__ biasU,
    const float* __restrict__ biasO, const float* __restrict__ biasC,
    float* __restrict__ cs, float* __restrict__ hs, int* __restrict__ bar) {
  extern __shared__ u16 lds[];
  u16* WfL = lds;                     // [32][768]
  u16* WuL = WfL + 32 * CD;
  u16* WoL = WuL + 32 * CD;
  u16* WcL = WoL + 32 * CD;           // [32][512]

  const int tid  = threadIdx.x;
  const int wave = tid >> 5;
  const int lane = tid & 31;
  const int half = lane >> 4;
  const int l16  = lane & 15;
  const int wg   = blockIdx.x;        // 0..31
  const int colblk = wg & 7;          // 8 col slabs of 32
  const int rowblk = wg >> 3;         // 4 row slabs of 64
  const int m0 = rowblk * 64 + (wave & 3) * 16;
  const int n0 = colblk * 32 + (wave >> 2) * 16;
  const int arow = m0 + l16;
  const int bcol = n0 + l16;
  const int nl   = (wave >> 2) * 16 + l16;   // column within LDS slab

  // ---- stage this WG's weight slab into LDS once (async path if available)
  const int colbase = colblk * 32;
  for (int idx = tid; idx < 32 * (CD / 8); idx += 256) {
    int col = idx / (CD / 8), ch = idx % (CD / 8);
    size_t go = (size_t)(colbase + col) * CD + ch * 8;
    size_t lo = (size_t)col * CD + ch * 8;
    stage16(WfT + go, WfL + lo);
    stage16(WuT + go, WuL + lo);
    stage16(WoT + go, WoL + lo);
  }
  for (int idx = tid; idx < 32 * (CW / 8); idx += 256) {
    int col = idx / (CW / 8), ch = idx % (CW / 8);
    stage16(WcT + (size_t)(colbase + col) * CW + ch * 8,
            WcL + (size_t)col * CW + ch * 8);
  }
  stage_wait();
  __syncthreads();

  const float bF = biasF[bcol], bU = biasU[bcol], bO = biasO[bcol], bC = biasC[bcol];

#pragma unroll 1
  for (int t = 0; t < CT; ++t) {
    const u16*  cbf_in  = (t & 1) ? cbfb : cbfa;
    const float* c32_in = (t & 1) ? c32b : c32a;
    float*      c32_out = (t & 1) ? c32a : c32b;
    u16*        cbf_out = (t & 1) ? cbfa : cbfb;
    const u16*  hbf_in  = t ? hsbf + (size_t)(t - 1) * CB * CH : h0;
    u16*        hbf_out = hsbf + (size_t)t * CB * CH;
    const u16*  xe_t    = xe + (size_t)t * CB * CE;

    v8f accF = {}, accU = {}, accO = {}, accC = {};

#pragma unroll
    for (int seg = 0; seg < 3; ++seg) {
      const u16* abase =
          (seg == 0 ? cbf_in : (seg == 1 ? hbf_in : xe_t)) + (size_t)arow * 256;
      for (int kc = 0; kc < 256; kc += 32) {
        V16BF A;                       // A operand: two 16B chunks per lane
        A.q[0] = *(const U128*)(abase + kc + half * 16);
        A.q[1] = *(const U128*)(abase + kc + half * 16 + 8);

        const int kg = seg * 256 + kc + half * 16;
        v16bf Bf = *(const v16bf*)(WfL + (size_t)nl * CD + kg);   // LDS
        v16bf Bu = *(const v16bf*)(WuL + (size_t)nl * CD + kg);
        v16bf Bo = *(const v16bf*)(WoL + (size_t)nl * CD + kg);
        accF = wmma_bf16(A.v, Bf, accF);
        accU = wmma_bf16(A.v, Bu, accU);
        accO = wmma_bf16(A.v, Bo, accO);
        if (seg > 0) {                 // c_new uses [h | x] only
          const int kcg = (seg - 1) * 256 + kc + half * 16;
          v16bf Bc = *(const v16bf*)(WcL + (size_t)nl * CW + kcg);
          accC = wmma_bf16(A.v, Bc, accC);
        }
      }
    }

    // Fused nonlinearities + state update; acc element r -> (m0+r+half*8, bcol)
#pragma unroll
    for (int r = 0; r < 8; ++r) {
      const int row = m0 + r + half * 8;
      const float f = sigm(accF[r] + bF);
      const float u = sigm(accU[r] + bU);
      const float o = sigm(accO[r] + bO);
      const float cn = tanhf(accC[r] + bC);
      const float cp = c32_in[(size_t)row * CH + bcol];
      const float ct = f * cp + u * cn;
      const float ht = o * tanhf(ct);
      c32_out[(size_t)row * CH + bcol] = ct;
      cbf_out[(size_t)row * CH + bcol] = f2bf(ct);
      hbf_out[(size_t)row * CH + bcol] = f2bf(ht);
      cs[(size_t)row * CT * CH + (size_t)t * CH + bcol] = ct;
      hs[(size_t)row * CT * CH + (size_t)t * CH + bcol] = ht;
    }

    if (t + 1 < CT) grid_barrier(bar, NWG * (t + 1));
  }
}

// ---------------------------------------------------------------------------
// Final projection: ys = hs @ Wy + by over all B*T rows at once.
// Grid: (V/128, B*T/16), 256 threads (8 waves, each a 16x16 N tile).
// ---------------------------------------------------------------------------
__global__ __launch_bounds__(256) void y_gemm(const u16* __restrict__ hsbf,
                                              const u16* __restrict__ WyT,
                                              const float* __restrict__ by,
                                              float* __restrict__ ys) {
  __shared__ u16 Atile[16 * 32];
  const int wave = threadIdx.x >> 5;
  const int lane = threadIdx.x & 31;
  const int half = lane >> 4;
  const int l16  = lane & 15;
  const int g0 = blockIdx.y * 16;                    // row tile in (t*B + b) space
  const int bcol = blockIdx.x * 128 + wave * 16 + l16;
  const int tid = threadIdx.x;

  v8f acc = {};
  for (int kc = 0; kc < CH; kc += 32) {
    __syncthreads();
    if (tid < 64) {                                  // stage 16x32 bf16 A tile
      const int row = tid >> 2, part = tid & 3;
      stage16(hsbf + (size_t)(g0 + row) * CH + kc + part * 8,
              Atile + row * 32 + part * 8);
    }
    stage_wait();
    __syncthreads();
    V16BF A;
    A.q[0] = *(const U128*)(Atile + l16 * 32 + half * 16);
    A.q[1] = *(const U128*)(Atile + l16 * 32 + half * 16 + 8);
    v16bf B = *(const v16bf*)(WyT + (size_t)bcol * CH + kc + half * 16);
    acc = wmma_bf16(A.v, B, acc);
  }

  const float bias = by[bcol];
#pragma unroll
  for (int r = 0; r < 8; ++r) {
    const int g = g0 + r + half * 8;                 // g = t*B + b
    const int b = g & (CB - 1);
    const int t = g >> 8;
    ys[(size_t)b * CT * CV + (size_t)t * CV + bcol] = acc[r] + bias;
  }
}

// ---------------------------------------------------------------------------
extern "C" void kernel_launch(void* const* d_in, const int* in_sizes, int n_in,
                              void* d_out, int out_size, void* d_ws, size_t ws_size,
                              hipStream_t stream) {
  (void)in_sizes; (void)n_in; (void)out_size; (void)ws_size;
  const int*   x     = (const int*)d_in[0];
  const float* embed = (const float*)d_in[1];
  const float* Wf = (const float*)d_in[2];  const float* bf = (const float*)d_in[3];
  const float* Wu = (const float*)d_in[4];  const float* bu = (const float*)d_in[5];
  const float* Wo = (const float*)d_in[6];  const float* bo = (const float*)d_in[7];
  const float* Wc = (const float*)d_in[8];  const float* bc = (const float*)d_in[9];
  const float* Wy = (const float*)d_in[10]; const float* by = (const float*)d_in[11];

  float* out = (float*)d_out;
  float* cs = out;                                   // [B,T,H]
  float* hs = out + (size_t)CB * CT * CH;            // [B,T,H]
  float* ys = out + (size_t)2 * CB * CT * CH;        // [B,T,V]

  char* w = (char*)d_ws;
  u16* xe    = (u16*)w;  w += (size_t)CT * CB * CE * 2;        // 64 MiB, [t][b][e]
  u16* hsbf  = (u16*)w;  w += (size_t)CT * CB * CH * 2;        // 64 MiB, [t][b][h]
  u16* h0    = (u16*)w;  w += (size_t)CB * CH * 2;
  u16* WfT   = (u16*)w;  w += (size_t)CH * CD * 2;
  u16* WuT   = (u16*)w;  w += (size_t)CH * CD * 2;
  u16* WoT   = (u16*)w;  w += (size_t)CH * CD * 2;
  u16* WcT   = (u16*)w;  w += (size_t)CH * CW * 2;
  u16* WyT   = (u16*)w;  w += (size_t)CV * CH * 2;
  float* c32a = (float*)w; w += (size_t)CB * CH * 4;
  float* c32b = (float*)w; w += (size_t)CB * CH * 4;
  u16* cbfa  = (u16*)w;  w += (size_t)CB * CH * 2;
  u16* cbfb  = (u16*)w;  w += (size_t)CB * CH * 2;
  int* bar   = (int*)w;  w += 256;

  prep_weights<<<(CD * CH + 255) / 256, 256, 0, stream>>>(
      Wf, Wu, Wo, Wc, Wy, WfT, WuT, WoT, WcT, WyT, c32a, cbfa, h0, bar);

  embed_gather<<<CB * CT, CE, 0, stream>>>(x, embed, xe);

  const size_t lds_bytes = (size_t)(3 * 32 * CD + 32 * CW) * sizeof(u16); // 176 KB
  lstm_scan<<<NWG, 256, lds_bytes, stream>>>(
      c32a, c32b, cbfa, cbfb, h0, hsbf, xe,
      WfT, WuT, WoT, WcT, bf, bu, bo, bc, cs, hs, bar);

  y_gemm<<<dim3(CV / 128, CB * CT / 16), 256, 0, stream>>>(hsbf, WyT, by, ys);
}
